// GNNModel_13804024889617
// MI455X (gfx1250) — compile-verified, matrix operand order checked by reading, not run on hardware
//
#include <hip/hip_runtime.h>
#include <hip/hip_bf16.h>
#include <stdint.h>

typedef __attribute__((ext_vector_type(16))) __bf16 v16bf;
typedef __attribute__((ext_vector_type(8)))  float  v8f;

#define D 128

// ---------------- helpers ----------------

__global__ void k_fill_f32(float* __restrict__ p, float v, int n) {
  for (int i = blockIdx.x * blockDim.x + threadIdx.x; i < n; i += gridDim.x * blockDim.x)
    p[i] = v;
}

__global__ void k_degree(const int* __restrict__ dst, float* __restrict__ deg, int E) {
  for (int i = blockIdx.x * blockDim.x + threadIdx.x; i < E; i += gridDim.x * blockDim.x)
    unsafeAtomicAdd(&deg[dst[i]], 1.0f);
}

__global__ void k_dinv(float* __restrict__ dinv, float* __restrict__ nself, int n) {
  for (int i = blockIdx.x * blockDim.x + threadIdx.x; i < n; i += gridDim.x * blockDim.x) {
    float r = rsqrtf(dinv[i]);   // dinv buffer currently holds deg
    dinv[i]  = r;
    nself[i] = r * r;
  }
}

__global__ void k_f32_to_bf16(const float* __restrict__ x, __bf16* __restrict__ y, int n) {
  for (int i = blockIdx.x * blockDim.x + threadIdx.x; i < n; i += gridDim.x * blockDim.x)
    y[i] = (__bf16)x[i];
}

// WT[n*128+k] = bf16(W[k*128+n])  (transpose so K is contiguous for fragment dword loads)
__global__ void k_weightT(const float* __restrict__ W, __bf16* __restrict__ WT) {
  int i = blockIdx.x * blockDim.x + threadIdx.x;
  if (i < D * D) {
    int n = i >> 7, k = i & 127;
    WT[i] = (__bf16)W[k * D + n];
  }
}

// ---------------- WMMA GEMM: C[MxD] = A[MxD](bf16) * W[DxD] ----------------
// WT is the bf16 transposed weight (n-major, K contiguous), staged into LDS with
// gfx1250 async-to-LDS DMA (ASYNCcnt). Block = 256 thr = 8 waves covering 128 rows.
// Wave w owns M-tile w; 8 N-tiles x 4 K-steps of v_wmma_f32_16x16x32_bf16,
// B fragments served from LDS.
template <bool HAS_BIAS, bool RELU, bool OUT_F32, bool OUT_BF16>
__global__ void k_gemm_bf16(const __bf16* __restrict__ A, const __bf16* __restrict__ WT,
                            float* __restrict__ Cf, __bf16* __restrict__ Cb,
                            const float* __restrict__ bias, int n_rows) {
  __shared__ __bf16 lw[D * D];               // 32 KB weight tile (whole matrix)
  {
    // Async DMA global -> LDS: 2048 x b128 chunks, 8 per thread, no VGPR staging.
    const uint32_t lbase = (uint32_t)(uintptr_t)(&lw[0]);
    const uint64_t gbase = (uint64_t)(uintptr_t)WT;
#pragma unroll
    for (int j = 0; j < 8; ++j) {
      const uint32_t i16 = (threadIdx.x + j * 256) * 16u;   // byte offset of chunk
      uint32_t la = lbase + i16;
      uint64_t ga = gbase + i16;
      asm volatile("global_load_async_to_lds_b128 %0, %1, off"
                   :: "v"(la), "v"(ga) : "memory");
    }
    asm volatile("s_wait_asynccnt 0" ::: "memory");
  }
  __syncthreads();

  const int wave = threadIdx.x >> 5;
  const int lane = threadIdx.x & 31;
  const int lmod = lane & 15;
  const int half = lane >> 4;

  const int mbase = blockIdx.x * 128 + wave * 16;
  int arow_idx = mbase + lmod;
  if (arow_idx >= n_rows) arow_idx = n_rows - 1;   // clamp (stores are guarded)
  const bool full_tile = (mbase + 16) <= n_rows;   // wave-uniform

  // A fragments: ISA 16-bit A layout. dword j of the 16-elem frag holds K pair:
  //   j<4 : K = 2j + 8*half ;  j>=4 : K = 16 + 2(j-4) + 8*half   (+ 32*kstep)
  union Frag { v16bf v; uint32_t u[8]; };
  Frag afr[4];
  const uint32_t* arow = (const uint32_t*)(A + (size_t)arow_idx * D);  // 64 dwords/row
#pragma unroll
  for (int ks = 0; ks < 4; ++ks) {
#pragma unroll
    for (int j = 0; j < 8; ++j) {
      int kb = (j < 4) ? (2 * j + 8 * half) : (16 + 2 * (j - 4) + 8 * half);
      afr[ks].u[j] = arow[(ks * 32 + kb) >> 1];
    }
  }

#pragma unroll 1
  for (int nt = 0; nt < 8; ++nt) {
    v8f acc = {};
    const uint32_t* brow = (const uint32_t*)(lw + (size_t)(nt * 16 + lmod) * D);
#pragma unroll
    for (int ks = 0; ks < 4; ++ks) {
      Frag bfr;
#pragma unroll
      for (int j = 0; j < 8; ++j) {
        int kb = (j < 4) ? (2 * j + 8 * half) : (16 + 2 * (j - 4) + 8 * half);
        bfr.u[j] = brow[(ks * 32 + kb) >> 1];
      }
      acc = __builtin_amdgcn_wmma_f32_16x16x32_bf16(
          false, afr[ks].v, false, bfr.v, (short)0, acc, false, false);
    }
    // C/D layout: lane -> N = lane%16 ; VGPR r -> M = r + 8*(lane/16)
    const int n = nt * 16 + lmod;
    float bv;
    if (HAS_BIAS) bv = bias[n];
    if (full_tile) {
#pragma unroll
      for (int r = 0; r < 8; ++r) {
        const int m = mbase + r + 8 * half;
        float v = acc[r];
        if (HAS_BIAS) v += bv;
        if (RELU) v = v > 0.0f ? v : 0.0f;
        if (OUT_F32)  Cf[(size_t)m * D + n] = v;
        if (OUT_BF16) Cb[(size_t)m * D + n] = (__bf16)v;
      }
    } else {
#pragma unroll
      for (int r = 0; r < 8; ++r) {
        const int m = mbase + r + 8 * half;
        if (m < n_rows) {
          float v = acc[r];
          if (HAS_BIAS) v += bv;
          if (RELU) v = v > 0.0f ? v : 0.0f;
          if (OUT_F32)  Cf[(size_t)m * D + n] = v;
          if (OUT_BF16) Cb[(size_t)m * D + n] = (__bf16)v;
        }
      }
    }
  }
}

// ---------------- aggregation ----------------

// agg[i,f] = hl[i,f]*nself[i] + b[f]   (self-loop term + bias folded into init)
__global__ void k_init_agg(const float* __restrict__ hl, const float* __restrict__ nself,
                           const float* __restrict__ bias, float* __restrict__ agg, int n) {
  for (int i = blockIdx.x * blockDim.x + threadIdx.x; i < n; i += gridDim.x * blockDim.x)
    agg[i] = hl[i] * nself[i >> 7] + bias[i & 127];
}

// one wave per edge: agg[dst] += hl[src] * dinv[src]*dinv[dst]
__global__ void k_scatter(const int* __restrict__ src, const int* __restrict__ dst,
                          const float* __restrict__ dinv, const float* __restrict__ hl,
                          float* __restrict__ agg, int E) {
  const int wpb = blockDim.x >> 5;
  const int lane = threadIdx.x & 31;
  int w = blockIdx.x * wpb + (threadIdx.x >> 5);
  const int stride = gridDim.x * wpb;
  for (int e = w; e < E; e += stride) {
    const int s = src[e], d = dst[e];
    const float nrm = dinv[s] * dinv[d];
    const float* hs = hl + (size_t)s * D;
    float* ad = agg + (size_t)d * D;
#pragma unroll
    for (int j = 0; j < 4; ++j) {
      int f = lane + 32 * j;
      unsafeAtomicAdd(&ad[f], hs[f] * nrm);
    }
  }
}

// h = relu(agg); write f32 (optional) + bf16 (next layer input)
template <bool OUT_F32>
__global__ void k_finish(const float* __restrict__ agg, float* __restrict__ hf,
                         __bf16* __restrict__ hb, int n) {
  for (int i = blockIdx.x * blockDim.x + threadIdx.x; i < n; i += gridDim.x * blockDim.x) {
    float v = agg[i];
    v = v > 0.0f ? v : 0.0f;
    if (OUT_F32) hf[i] = v;
    hb[i] = (__bf16)v;
  }
}

// ---------------- anomaly head: score[i] = h[i,:] . Wa + ba ----------------
__global__ void k_anomaly(const float* __restrict__ h, const float* __restrict__ Wa,
                          const float* __restrict__ ba, float* __restrict__ out, int n) {
  const int wpb = blockDim.x >> 5;
  const int lane = threadIdx.x & 31;
  const int node = blockIdx.x * wpb + (threadIdx.x >> 5);
  if (node >= n) return;
  const float* hr = h + (size_t)node * D;
  float s = 0.0f;
#pragma unroll
  for (int j = 0; j < 4; ++j) {
    int f = lane + 32 * j;
    s += hr[f] * Wa[f];
  }
#pragma unroll
  for (int m = 16; m > 0; m >>= 1) s += __shfl_xor(s, m, 32);
  if (lane == 0) out[node] = s + ba[0];
}

// ---------------- host orchestration ----------------

extern "C" void kernel_launch(void* const* d_in, const int* in_sizes, int n_in,
                              void* d_out, int out_size, void* d_ws, size_t ws_size,
                              hipStream_t stream) {
  const float* x   = (const float*)d_in[0];
  const int*   ei  = (const int*)d_in[1];
  const float* W0  = (const float*)d_in[2];
  const float* b0  = (const float*)d_in[3];
  const float* W1  = (const float*)d_in[4];
  const float* b1  = (const float*)d_in[5];
  const float* W2  = (const float*)d_in[6];
  const float* b2  = (const float*)d_in[7];
  const float* Wd0 = (const float*)d_in[8];
  const float* bd0 = (const float*)d_in[9];
  const float* Wd1 = (const float*)d_in[10];
  const float* bd1 = (const float*)d_in[11];
  const float* Wa  = (const float*)d_in[12];
  const float* ba  = (const float*)d_in[13];

  const int N = in_sizes[0] / D;
  const int E = in_sizes[1] / 2;
  const int* esrc = ei;
  const int* edst = ei + E;
  const int NF = N * D;

  // workspace layout (~78 MB)
  char* ws = (char*)d_ws;
  size_t off = 0;
  auto alloc = [&](size_t bytes) -> void* {
    void* p = ws + off;
    off += (bytes + 255) & ~(size_t)255;
    return p;
  };
  float*  dinv  = (float*)alloc((size_t)N * 4);       // holds deg first
  float*  nself = (float*)alloc((size_t)N * 4);
  __bf16* WT    = (__bf16*)alloc((size_t)5 * D * D * 2);
  __bf16* B0    = (__bf16*)alloc((size_t)NF * 2);     // bf16 activations (GEMM input)
  float*  B1    = (float*)alloc((size_t)NF * 4);      // f32 GEMM output (and bf16 't' later)

  float* outH   = (float*)d_out;                      // [N,128] embeddings
  float* rec    = outH + (size_t)NF;                  // [N,128] reconstruction
  float* ascore = rec + (size_t)NF;                   // [N]     anomaly scores
  float* agg    = rec;                                // reuse rec region as agg scratch

  const int T = 256;
  const int gN  = (N + T - 1) / T;
  const int gE  = (E + T - 1) / T;
  const int gNF = (NF + T - 1) / T;
  const int gGemm = (N + 127) / 128;
  const int gWaveE = (E + 7) / 8;   // 8 waves per block
  const int gWaveN = (N + 7) / 8;

  // degrees + normalization
  k_fill_f32<<<gN, T, 0, stream>>>(dinv, 1.0f, N);
  k_degree<<<gE, T, 0, stream>>>(edst, dinv, E);
  k_dinv<<<gN, T, 0, stream>>>(dinv, nself, N);

  // precision conversion
  k_f32_to_bf16<<<gNF, T, 0, stream>>>(x, B0, NF);
  const float* Ws[5] = {W0, W1, W2, Wd0, Wd1};
  for (int w = 0; w < 5; ++w)
    k_weightT<<<(D * D + T - 1) / T, T, 0, stream>>>(Ws[w], WT + (size_t)w * D * D);

  // 3 GCN conv layers
  const float* bs[3] = {b0, b1, b2};
  for (int l = 0; l < 3; ++l) {
    k_gemm_bf16<false, false, true, false><<<gGemm, T, 0, stream>>>(
        B0, WT + (size_t)l * D * D, B1, nullptr, nullptr, N);
    k_init_agg<<<gNF, T, 0, stream>>>(B1, nself, bs[l], agg, NF);
    k_scatter<<<gWaveE, T, 0, stream>>>(esrc, edst, dinv, B1, agg, E);
    if (l == 2)
      k_finish<true><<<gNF, T, 0, stream>>>(agg, outH, B0, NF);
    else
      k_finish<false><<<gNF, T, 0, stream>>>(agg, nullptr, B0, NF);
  }

  // decoder: t = relu(h@Wd0+bd0) (bf16, into B1's memory); rec = t@Wd1+bd1
  __bf16* tB = (__bf16*)B1;
  k_gemm_bf16<true, true, false, true><<<gGemm, T, 0, stream>>>(
      B0, WT + (size_t)3 * D * D, nullptr, tB, bd0, N);
  k_gemm_bf16<true, false, true, false><<<gGemm, T, 0, stream>>>(
      tB, WT + (size_t)4 * D * D, rec, nullptr, bd1, N);

  // anomaly head
  k_anomaly<<<gWaveN, T, 0, stream>>>(outH, Wa, ba, ascore, N);
}
